// Det3DTransformer_67233418051808
// MI455X (gfx1250) — matrix operand order repaired and checked by reading, not compiled, not run on hardware
//
#include <hip/hip_runtime.h>
#include <hip/hip_bf16.h>
#include <math.h>

// ---------------------------------------------------------------------------
// Det3D transformer decoder for MI455X (gfx1250, wave32, WMMA).
// Dense GEMMs: v_wmma_f32_16x16x32_bf16 on pre-converted bf16 operands,
// 32(M) x 64(N) tile per wave (8 WMMAs / K-step, B fragments reused x2,
// all fragment loads issued before the WMMA burst so loads overlap WMMAs).
// Attention softmax / LayerNorm / deformable sampling: wave32 VALU kernels,
// bf16 in/out wherever the next consumer is a GEMM.
// ---------------------------------------------------------------------------

#define BSZ     8
#define NQL     550
#define DMODEL  256
#define NHEAD   8
#define NLVL    4
#define NPTS    4
#define NLAYERS 6
#define DFFN    1024
#define STOT    10200   // 48*160 + 24*80 + 12*40 + 6*20
#define DEPHW   1920    // 24*80
#define DH      32      // DMODEL / NHEAD
#define MROWS   (BSZ * NQL)          // 4400 logical query rows
#define MPAD    4416                 // padded to a multiple of 32 for GEMM tiles

typedef __attribute__((ext_vector_type(16))) __bf16          v16bf;
typedef __attribute__((ext_vector_type(16))) unsigned short  v16u;
typedef __attribute__((ext_vector_type(8)))  unsigned short  v8u;
typedef __attribute__((ext_vector_type(8)))  float           v8f;

// round-to-nearest-even fp32 -> bf16 bits
static __device__ __forceinline__ unsigned short f2bfbits(float x) {
    union { float f; unsigned int u; } v; v.f = x;
    unsigned int r = v.u + 0x7fffu + ((v.u >> 16) & 1u);
    return (unsigned short)(r >> 16);
}
static __device__ __forceinline__ float bf2f(unsigned short h) {
    union { unsigned int u; float f; } v; v.u = ((unsigned int)h) << 16;
    return v.f;
}

// A fragment (16x32 bf16): elements 0..7 <- p[0..7], 8..15 <- p[16..23]
static __device__ __forceinline__ v16bf ldfragA(const unsigned short* p) {
    v8u lo = *(const v8u*)(p);
    v8u hi = *(const v8u*)(p + 16);
    return __builtin_bit_cast(v16bf, __builtin_shufflevector(
        lo, hi, 0, 1, 2, 3, 4, 5, 6, 7, 8, 9, 10, 11, 12, 13, 14, 15));
}
// B fragment (32x16 bf16): elements 0..7 <- p[0..7], 8..15 <- p[16..23]
static __device__ __forceinline__ v16bf ldfragB(const unsigned short* p) {
    v8u lo = *(const v8u*)(p);
    v8u hi = *(const v8u*)(p + 16);
    return __builtin_bit_cast(v16bf, __builtin_shufflevector(
        lo, hi, 0, 1, 2, 3, 4, 5, 6, 7, 8, 9, 10, 11, 12, 13, 14, 15));
}

// ---------------------------------------------------------------------------
// bulk fp32 -> bf16 conversion (one-time per tensor)
// ---------------------------------------------------------------------------
__global__ void cvt_bf16_kernel(const float* __restrict__ in,
                                unsigned short* __restrict__ out, int n)
{
    int i = blockIdx.x * blockDim.x + threadIdx.x;
    if (i < n) out[i] = f2bfbits(in[i]);
}

// tgt -> x (f32) and xbf (bf16)
__global__ void init_x_kernel(const float* __restrict__ tgt,
                              float* __restrict__ x,
                              unsigned short* __restrict__ xbf, int n)
{
    int i = blockIdx.x * blockDim.x + threadIdx.x;
    if (i < n) { float v = tgt[i]; x[i] = v; xbf[i] = f2bfbits(v); }
}

// depth_pos_embed [B, D, 1920] -> dkv bf16 [B, 1920, D]
__global__ void transpose_dkv_kernel(const float* __restrict__ dpe,
                                     unsigned short* __restrict__ dkv, int total)
{
    int i = blockIdx.x * blockDim.x + threadIdx.x;
    if (i >= total) return;
    int d = i % DMODEL;
    int rest = i / DMODEL;
    int sidx = rest % DEPHW;
    int b = rest / DEPHW;
    dkv[i] = f2bfbits(dpe[((size_t)b * DMODEL + d) * DEPHW + sidx]);
}

// ---------------------------------------------------------------------------
// GEMM: C[M,N] = act( A[M,K] @ W[N,K]^T + bias[N] ), A/W bf16, f32 accumulate.
// One wave per 32(M) x 64(N) tile: 2 A frags + 4 B frags -> 8 WMMAs / K-step.
// All loads for a K-step are issued before the WMMA burst (overlap).
// Optional f32 (Cf) and/or bf16 (Cb) outputs.
// Requires M%32==0, N%64==0, K%32==0 (guaranteed via MPAD row padding).
//
// Fragment layouts per CDNA5 ISA 7.12.2 (wave32):
//   A (16x32 bf16): lane l -> row m = l&15; group g = l>>4;
//       elements 0..7 -> K = 8g..8g+7 ; elements 8..15 -> K = 16+8g..16+8g+7
//   B (32x16 bf16): lane l -> col n = l&15; same K order as A
//   D (16x16 f32):  lane l -> col n = l&15; vgpr r -> row m = r + 8g
// ---------------------------------------------------------------------------
__global__ void __launch_bounds__(32)
gemm_bf16_wmma(const unsigned short* __restrict__ A,
               const unsigned short* __restrict__ W,
               const float* __restrict__ bias,
               float* __restrict__ Cf, unsigned short* __restrict__ Cb,
               int M, int N, int K, int act)
{
    const int lane = threadIdx.x;
    const int g    = lane >> 4;
    const int lr   = lane & 15;
    const int m0   = blockIdx.y * 32;
    const int n0   = blockIdx.x * 64;

    const unsigned short* arow0 = A + (size_t)(m0 + lr) * K + 8 * g;
    const unsigned short* arow1 = arow0 + (size_t)16 * K;
    const unsigned short* wrow0 = W + (size_t)(n0 + lr) * K + 8 * g;

    v8f acc[2][4];
#pragma unroll
    for (int i = 0; i < 2; ++i)
#pragma unroll
        for (int t = 0; t < 4; ++t) acc[i][t] = v8f{};

    for (int kk = 0; kk < K; kk += 32) {
        // issue ALL fragment loads first, then the WMMA burst: lets the
        // scheduler wait on partial loadcnt and overlap loads with WMMAs
        v16bf a0 = ldfragA(arow0 + kk);
        v16bf a1 = ldfragA(arow1 + kk);
        v16bf b[4];
#pragma unroll
        for (int t = 0; t < 4; ++t)
            b[t] = ldfragB(wrow0 + (size_t)(t * 16) * K + kk);
#pragma unroll
        for (int t = 0; t < 4; ++t)
            acc[0][t] = __builtin_amdgcn_wmma_f32_16x16x32_bf16(
                false, a0, false, b[t], (short)0, acc[0][t], false, false);
#pragma unroll
        for (int t = 0; t < 4; ++t)
            acc[1][t] = __builtin_amdgcn_wmma_f32_16x16x32_bf16(
                false, a1, false, b[t], (short)0, acc[1][t], false, false);
    }

#pragma unroll
    for (int i = 0; i < 2; ++i) {
#pragma unroll
        for (int t = 0; t < 4; ++t) {
            const int n  = n0 + t * 16 + lr;
            const float bb = bias[n];
#pragma unroll
            for (int r = 0; r < 8; ++r) {
                const int m = m0 + i * 16 + 8 * g + r;
                float v = acc[i][t][r] + bb;
                if (act) v = fmaxf(v, 0.f);
                if (Cf) Cf[(size_t)m * N + n] = v;
                if (Cb) Cb[(size_t)m * N + n] = f2bfbits(v);
            }
        }
    }
}

// ---------------------------------------------------------------------------
// Softmax attention: one wave per (b, h, q). bf16 Q/K/V in, bf16 out.
// Scores in LDS, lane = output dim (dh == wave32).
// ---------------------------------------------------------------------------
__global__ void __launch_bounds__(32)
attn_kernel(const unsigned short* __restrict__ Q, int qStride,
            const unsigned short* __restrict__ Km, int kStride,
            const unsigned short* __restrict__ Vm, int vStride,
            unsigned short* __restrict__ O, int oStride,
            int Lq, int Lk, float scale)
{
    __shared__ float sc[DEPHW];   // max Lk = 1920
    __shared__ float qs[DH];
    const int lane = threadIdx.x;
    const int qi = blockIdx.x, h = blockIdx.y, b = blockIdx.z;

    const unsigned short* qp = Q + (size_t)(b * Lq + qi) * qStride + h * DH;
    qs[lane] = bf2f(qp[lane]) * scale;
    __syncthreads();

    float lmax = -3.0e38f;
    for (int j = lane; j < Lk; j += 32) {
        const unsigned short* kp = Km + (size_t)(b * Lk + j) * kStride + h * DH;
        float s = 0.f;
#pragma unroll
        for (int d = 0; d < DH; ++d) s += qs[d] * bf2f(kp[d]);
        sc[j] = s;
        lmax = fmaxf(lmax, s);
    }
    for (int o = 16; o > 0; o >>= 1) lmax = fmaxf(lmax, __shfl_xor(lmax, o, 32));
    __syncthreads();

    float lsum = 0.f;
    for (int j = lane; j < Lk; j += 32) {
        float p = __expf(sc[j] - lmax);
        sc[j] = p;
        lsum += p;
    }
    for (int o = 16; o > 0; o >>= 1) lsum += __shfl_xor(lsum, o, 32);
    __syncthreads();

    float acc = 0.f;
    for (int j = 0; j < Lk; ++j)
        acc += sc[j] * bf2f(Vm[(size_t)(b * Lk + j) * vStride + h * DH + lane]);
    O[(size_t)(b * Lq + qi) * oStride + h * DH + lane] = f2bfbits(acc / lsum);
}

// ---------------------------------------------------------------------------
// x = LayerNorm(x + t2) * g + b ; also writes bf16 copy of x for the GEMMs.
// ---------------------------------------------------------------------------
__global__ void __launch_bounds__(DMODEL)
add_ln_kernel(float* __restrict__ x, unsigned short* __restrict__ xbf,
              const float* __restrict__ t2,
              const float* __restrict__ gam, const float* __restrict__ bet)
{
    __shared__ float red[DMODEL];
    const int tid = threadIdx.x;
    const size_t row = blockIdx.x;
    float v = x[row * DMODEL + tid] + t2[row * DMODEL + tid];

    red[tid] = v; __syncthreads();
    for (int st = DMODEL / 2; st > 0; st >>= 1) {
        if (tid < st) red[tid] += red[tid + st];
        __syncthreads();
    }
    const float mu = red[0] * (1.f / DMODEL);
    __syncthreads();

    const float c = v - mu;
    red[tid] = c * c; __syncthreads();
    for (int st = DMODEL / 2; st > 0; st >>= 1) {
        if (tid < st) red[tid] += red[tid + st];
        __syncthreads();
    }
    const float var = red[0] * (1.f / DMODEL);
    const float o = c * rsqrtf(var + 1e-5f) * gam[tid] + bet[tid];
    x[row * DMODEL + tid]   = o;
    xbf[row * DMODEL + tid] = f2bfbits(o);
}

// ---------------------------------------------------------------------------
// Deformable attention sampling. One wave per (b, h, q); lane = channel d.
// value tensor is bf16 (halves gather bandwidth). mask_flatten is all-false.
// ---------------------------------------------------------------------------
__global__ void __launch_bounds__(32)
msda_sample_kernel(const float* __restrict__ off,          // [MPAD, 256] (h,l,p,2)
                   const float* __restrict__ awr,          // [MPAD, 128] (h,l,p)
                   const float* __restrict__ refp,         // [B, NQ, 2]
                   const float* __restrict__ vr,           // [B, NL, 2]
                   const unsigned short* __restrict__ val, // [B*S, 256] bf16
                   unsigned short* __restrict__ outp)      // [MPAD, 256] bf16
{
    const int d  = threadIdx.x;
    const int qi = blockIdx.x, h = blockIdx.y, b = blockIdx.z;
    const int lvlH[NLVL] = {48, 24, 12, 6};
    const int lvlW[NLVL] = {160, 80, 40, 20};
    const int lvlS[NLVL] = {0, 7680, 9600, 10080};

    const size_t qrow = (size_t)b * NQL + qi;

    const float* ar = awr + qrow * (NHEAD * NLVL * NPTS) + h * (NLVL * NPTS);
    float aw[NLVL * NPTS];
    float mx = -3.0e38f;
#pragma unroll
    for (int i = 0; i < NLVL * NPTS; ++i) mx = fmaxf(mx, ar[i]);
    float s = 0.f;
#pragma unroll
    for (int i = 0; i < NLVL * NPTS; ++i) { aw[i] = __expf(ar[i] - mx); s += aw[i]; }
    const float inv = 1.f / s;

    const float rx = refp[qrow * 2 + 0];
    const float ry = refp[qrow * 2 + 1];
    const float* op = off + qrow * DMODEL + h * (NLVL * NPTS * 2);

    float acc = 0.f;
    for (int l = 0; l < NLVL; ++l) {
        const int Hl = lvlH[l], Wl = lvlW[l], st = lvlS[l];
        const float bx = rx * vr[(b * NLVL + l) * 2 + 0];
        const float by = ry * vr[(b * NLVL + l) * 2 + 1];
#pragma unroll
        for (int p = 0; p < NPTS; ++p) {
            const float ox = op[l * (NPTS * 2) + p * 2 + 0];
            const float oy = op[l * (NPTS * 2) + p * 2 + 1];
            const float px = (bx + ox / (float)Wl) * (float)Wl - 0.5f;
            const float py = (by + oy / (float)Hl) * (float)Hl - 0.5f;
            const float wgt = aw[l * NPTS + p] * inv;
            const float x0 = floorf(px), y0 = floorf(py);
            float sv = 0.f;
#pragma unroll
            for (int dy = 0; dy < 2; ++dy) {
#pragma unroll
                for (int dx = 0; dx < 2; ++dx) {
                    const float xi = x0 + dx, yi = y0 + dy;
                    if (xi >= 0.f && xi < (float)Wl && yi >= 0.f && yi < (float)Hl) {
                        const float w = (1.f - fabsf(px - xi)) * (1.f - fabsf(py - yi));
                        sv += w * bf2f(val[((size_t)b * STOT + st + (int)yi * Wl + (int)xi)
                                           * DMODEL + h * DH + d]);
                    }
                }
            }
            acc += wgt * sv;
        }
    }
    outp[qrow * DMODEL + h * DH + d] = f2bfbits(acc);
}

// d_out = concat(hs flat, ref_points flat, ref_points flat)
__global__ void write_out_kernel(const float* __restrict__ x,
                                 const float* __restrict__ refp,
                                 float* __restrict__ out, int nx, int nr)
{
    int i = blockIdx.x * blockDim.x + threadIdx.x;
    if (i >= nx + 2 * nr) return;
    if (i < nx) out[i] = x[i];
    else { int j = i - nx; out[i] = refp[j % nr]; }
}

// ---------------------------------------------------------------------------
static inline void launch_gemm(const unsigned short* A, const unsigned short* W,
                               const float* bias, float* Cf, unsigned short* Cb,
                               int M, int N, int K, int act, hipStream_t stream)
{
    dim3 grid(N / 64, M / 32);
    gemm_bf16_wmma<<<grid, dim3(32), 0, stream>>>(A, W, bias, Cf, Cb, M, N, K, act);
}

static inline void launch_cvt(const float* in, unsigned short* out, size_t n,
                              hipStream_t stream)
{
    cvt_bf16_kernel<<<(unsigned)((n + 255) / 256), 256, 0, stream>>>(in, out, (int)n);
}

extern "C" void kernel_launch(void* const* d_in, const int* in_sizes, int n_in,
                              void* d_out, int out_size, void* d_ws, size_t ws_size,
                              hipStream_t stream)
{
    (void)in_sizes; (void)n_in; (void)out_size; (void)ws_size;

    // inputs in setup_inputs() dict order
    const float* tgt      = (const float*)d_in[0];
    const float* refp     = (const float*)d_in[1];
    const float* memory   = (const float*)d_in[2];
    const float* vratios  = (const float*)d_in[5];
    const float* dpe      = (const float*)d_in[7];

    const float* dep_Wqkv = (const float*)d_in[8];
    const float* dep_bqkv = (const float*)d_in[9];
    const float* dep_Wo   = (const float*)d_in[10];
    const float* dep_bo   = (const float*)d_in[11];
    const float* sa_Wqkv  = (const float*)d_in[12];
    const float* sa_bqkv  = (const float*)d_in[13];
    const float* sa_Wo    = (const float*)d_in[14];
    const float* sa_bo    = (const float*)d_in[15];
    const float* ms_Wv    = (const float*)d_in[16];
    const float* ms_bv    = (const float*)d_in[17];
    const float* ms_Woff  = (const float*)d_in[18];
    const float* ms_boff  = (const float*)d_in[19];
    const float* ms_Wattn = (const float*)d_in[20];
    const float* ms_battn = (const float*)d_in[21];
    const float* ms_Wout  = (const float*)d_in[22];
    const float* ms_bout  = (const float*)d_in[23];
    const float* W1       = (const float*)d_in[24];
    const float* b1       = (const float*)d_in[25];
    const float* W2       = (const float*)d_in[26];
    const float* b2       = (const float*)d_in[27];
    const float* lndep_g  = (const float*)d_in[28];
    const float* lndep_b  = (const float*)d_in[29];
    const float* ln2_g    = (const float*)d_in[30];
    const float* ln2_b    = (const float*)d_in[31];
    const float* ln1_g    = (const float*)d_in[32];
    const float* ln1_b    = (const float*)d_in[33];
    const float* ln3_g    = (const float*)d_in[34];
    const float* ln3_b    = (const float*)d_in[35];

    const int M  = MROWS;            // 4400 logical rows
    const int MP = MPAD;             // 4416 padded rows for 32-row GEMM tiles
    const int Mk = BSZ * DEPHW;      // 15360 (already %32)
    const int Mv = BSZ * STOT;       // 81600 (already %32)
    const float scale = 1.0f / sqrtf((float)DH);

    // ---- workspace layout ----
    // f32 region first, then bf16 (ushort) region.
    // Buffers touched by 32-row GEMM tiles are padded to MP rows; the 16
    // padding rows only ever hold garbage that is never consumed.
    float* wsf = (float*)d_ws;
    size_t curf = 0;
    auto allocF = [&](size_t n) { float* p = wsf + curf; curf += n; return p; };
    float* x      = allocF((size_t)M  * DMODEL);    // hidden state (f32 master)
    float* bufQ   = allocF((size_t)MP * DMODEL);    // msda offsets
    float* bufAW  = allocF((size_t)MP * 128);       // msda attn logits
    float* bufT2  = allocF((size_t)MP * DMODEL);    // projection output (residual)

    unsigned short* wsu = (unsigned short*)(wsf + curf);
    size_t curu = 0;
    auto allocU = [&](size_t n) { unsigned short* p = wsu + curu; curu += n; return p; };
    unsigned short* xbf   = allocU((size_t)MP * DMODEL);
    unsigned short* qbf   = allocU((size_t)MP * DMODEL);      // dep-attention Q
    unsigned short* dkvbf = allocU((size_t)Mk * DMODEL);      // depth KV source
    unsigned short* membf = allocU((size_t)Mv * DMODEL);      // memory (constant)
    unsigned short* kvbf  = allocU((size_t)Mk * 2 * DMODEL);  // dep KV / self QKV
    unsigned short* valbf = allocU((size_t)Mv * DMODEL);      // msda value
    unsigned short* tbf   = allocU((size_t)MP * DMODEL);      // attn/msda output
    unsigned short* hbf   = allocU((size_t)MP * DFFN);        // FFN hidden

    // weights, bf16, all layers at once
    unsigned short* wb_depQkv = allocU((size_t)NLAYERS * 3 * DMODEL * DMODEL);
    unsigned short* wb_depWo  = allocU((size_t)NLAYERS * DMODEL * DMODEL);
    unsigned short* wb_saQkv  = allocU((size_t)NLAYERS * 3 * DMODEL * DMODEL);
    unsigned short* wb_saWo   = allocU((size_t)NLAYERS * DMODEL * DMODEL);
    unsigned short* wb_msWv   = allocU((size_t)NLAYERS * DMODEL * DMODEL);
    unsigned short* wb_msWoff = allocU((size_t)NLAYERS * DMODEL * DMODEL);
    unsigned short* wb_msWat  = allocU((size_t)NLAYERS * 128 * DMODEL);
    unsigned short* wb_msWout = allocU((size_t)NLAYERS * DMODEL * DMODEL);
    unsigned short* wb_W1     = allocU((size_t)NLAYERS * DFFN * DMODEL);
    unsigned short* wb_W2     = allocU((size_t)NLAYERS * DMODEL * DFFN);

    // ---- one-time conversions ----
    launch_cvt(dep_Wqkv, wb_depQkv, (size_t)NLAYERS * 3 * DMODEL * DMODEL, stream);
    launch_cvt(dep_Wo,   wb_depWo,  (size_t)NLAYERS * DMODEL * DMODEL,     stream);
    launch_cvt(sa_Wqkv,  wb_saQkv,  (size_t)NLAYERS * 3 * DMODEL * DMODEL, stream);
    launch_cvt(sa_Wo,    wb_saWo,   (size_t)NLAYERS * DMODEL * DMODEL,     stream);
    launch_cvt(ms_Wv,    wb_msWv,   (size_t)NLAYERS * DMODEL * DMODEL,     stream);
    launch_cvt(ms_Woff,  wb_msWoff, (size_t)NLAYERS * DMODEL * DMODEL,     stream);
    launch_cvt(ms_Wattn, wb_msWat,  (size_t)NLAYERS * 128 * DMODEL,        stream);
    launch_cvt(ms_Wout,  wb_msWout, (size_t)NLAYERS * DMODEL * DMODEL,     stream);
    launch_cvt(W1,       wb_W1,     (size_t)NLAYERS * DFFN * DMODEL,       stream);
    launch_cvt(W2,       wb_W2,     (size_t)NLAYERS * DMODEL * DFFN,       stream);
    launch_cvt(memory,   membf,     (size_t)Mv * DMODEL,                   stream);

    {
        int n = M * DMODEL;
        init_x_kernel<<<(n + 255) / 256, 256, 0, stream>>>(tgt, x, xbf, n);
        int t = Mk * DMODEL;
        transpose_dkv_kernel<<<(t + 255) / 256, 256, 0, stream>>>(dpe, dkvbf, t);
    }

    const dim3 attnGrid(NQL, NHEAD, BSZ);
    const dim3 lnGrid(M);

    for (int L = 0; L < NLAYERS; ++L) {
        const size_t wQKV = (size_t)L * 3 * DMODEL * DMODEL;
        const size_t bQKV = (size_t)L * 3 * DMODEL;
        const size_t wDD  = (size_t)L * DMODEL * DMODEL;
        const size_t bD   = (size_t)L * DMODEL;

        // ---- depth cross-attention ----
        launch_gemm(xbf, wb_depQkv + wQKV, dep_bqkv + bQKV,
                    nullptr, qbf, MP, DMODEL, DMODEL, 0, stream);
        launch_gemm(dkvbf, wb_depQkv + wQKV + (size_t)DMODEL * DMODEL,
                    dep_bqkv + bQKV + DMODEL,
                    nullptr, kvbf, Mk, 2 * DMODEL, DMODEL, 0, stream);
        attn_kernel<<<attnGrid, 32, 0, stream>>>(
            qbf, DMODEL, kvbf, 2 * DMODEL, kvbf + DMODEL, 2 * DMODEL,
            tbf, DMODEL, NQL, DEPHW, scale);
        launch_gemm(tbf, wb_depWo + wDD, dep_bo + bD,
                    bufT2, nullptr, MP, DMODEL, DMODEL, 0, stream);
        add_ln_kernel<<<lnGrid, DMODEL, 0, stream>>>(x, xbf, bufT2,
                                                     lndep_g + bD, lndep_b + bD);

        // ---- self-attention ----
        launch_gemm(xbf, wb_saQkv + wQKV, sa_bqkv + bQKV,
                    nullptr, kvbf, MP, 3 * DMODEL, DMODEL, 0, stream);
        attn_kernel<<<attnGrid, 32, 0, stream>>>(
            kvbf, 3 * DMODEL, kvbf + DMODEL, 3 * DMODEL, kvbf + 2 * DMODEL, 3 * DMODEL,
            tbf, DMODEL, NQL, NQL, scale);
        launch_gemm(tbf, wb_saWo + wDD, sa_bo + bD,
                    bufT2, nullptr, MP, DMODEL, DMODEL, 0, stream);
        add_ln_kernel<<<lnGrid, DMODEL, 0, stream>>>(x, xbf, bufT2,
                                                     ln2_g + bD, ln2_b + bD);

        // ---- deformable cross-attention ----
        launch_gemm(membf, wb_msWv + wDD, ms_bv + bD,
                    nullptr, valbf, Mv, DMODEL, DMODEL, 0, stream);
        launch_gemm(xbf, wb_msWoff + wDD, ms_boff + bD,
                    bufQ, nullptr, MP, DMODEL, DMODEL, 0, stream);
        launch_gemm(xbf, wb_msWat + (size_t)L * 128 * DMODEL, ms_battn + (size_t)L * 128,
                    bufAW, nullptr, MP, 128, DMODEL, 0, stream);
        msda_sample_kernel<<<attnGrid, 32, 0, stream>>>(
            bufQ, bufAW, refp, vratios, valbf, tbf);
        launch_gemm(tbf, wb_msWout + wDD, ms_bout + bD,
                    bufT2, nullptr, MP, DMODEL, DMODEL, 0, stream);
        add_ln_kernel<<<lnGrid, DMODEL, 0, stream>>>(x, xbf, bufT2,
                                                     ln1_g + bD, ln1_b + bD);

        // ---- FFN ----
        launch_gemm(xbf, wb_W1 + (size_t)L * DFFN * DMODEL, b1 + (size_t)L * DFFN,
                    nullptr, hbf, MP, DFFN, DMODEL, 1, stream);
        launch_gemm(hbf, wb_W2 + (size_t)L * DMODEL * DFFN, b2 + bD,
                    bufT2, nullptr, MP, DMODEL, DFFN, 0, stream);
        add_ln_kernel<<<lnGrid, DMODEL, 0, stream>>>(x, xbf, bufT2,
                                                     ln3_g + bD, ln3_b + bD);
    }

    const int nx = M * DMODEL;     // 1,126,400
    const int nr = M * 2;          // 8,800
    write_out_kernel<<<(nx + 2 * nr + 255) / 256, 256, 0, stream>>>(
        x, refp, (float*)d_out, nx, nr);
}
